// Decoder_40269613367326
// MI455X (gfx1250) — compile-verified, matrix operand order checked by reading, not run on hardware
//
#include <hip/hip_runtime.h>
#include <hip/hip_bf16.h>

// ---------------------------------------------------------------------------
// MI455X (gfx1250) transformer decoder layer.
// All GEMMs + attention matmuls on V_WMMA_F32_16X16X32_BF16 (wave32).
// Weights pre-transposed once so GEMM LDS staging is pure 16B transactions;
// async global->LDS (ASYNCcnt) double-buffered pipeline when available.
// ---------------------------------------------------------------------------

typedef __bf16 bf16x4  __attribute__((ext_vector_type(4)));
typedef __bf16 bf16x8  __attribute__((ext_vector_type(8)));
typedef __bf16 bf16x16 __attribute__((ext_vector_type(16)));
typedef float  v8f     __attribute__((ext_vector_type(8)));
typedef int    v4i     __attribute__((ext_vector_type(4)));

union Frag { bf16x16 v; bf16x8 h[2]; };

#define AS1 __attribute__((address_space(1)))
#define AS3 __attribute__((address_space(3)))

#if __has_builtin(__builtin_amdgcn_global_load_async_to_lds_b128) && \
    __has_builtin(__builtin_amdgcn_s_wait_asynccnt)
#define USE_ASYNC_LDS 1
#else
#define USE_ASYNC_LDS 0
#endif

#if USE_ASYNC_LDS
// generic -> AS-qualified int4* cast chains (addrspacecast + reinterpret)
__device__ __forceinline__ AS1 v4i* as_global_v4i(const void* p) {
    return (AS1 v4i*)(AS1 void*)(void*)p;
}
__device__ __forceinline__ AS3 v4i* as_local_v4i(void* p) {
    return (AS3 v4i*)(AS3 void*)p;
}
#endif

__device__ __forceinline__ v8f wmma_bf16(const bf16x16& a, const bf16x16& b, v8f c) {
    // D = A(16x32 bf16) * B(32x16 bf16) + C(16x16 f32)
    return __builtin_amdgcn_wmma_f32_16x16x32_bf16(
        /*neg_a=*/false, a, /*neg_b=*/false, b,
        /*c_mod=*/(short)0, c, /*reuse_a=*/false, /*reuse_b=*/false);
}

// ---------------------------------------------------------------------------
// fp32 -> bf16 elementwise (n multiple of 1024)
// ---------------------------------------------------------------------------
__global__ __launch_bounds__(256) void cvt_f32_bf16(const float* __restrict__ in,
                                                    __bf16* __restrict__ out, int n) {
    int i = (blockIdx.x * 256 + threadIdx.x) * 4;
    if (i + 3 < n) {
        float4 v = *(const float4*)(in + i);
        bf16x4 o;
        o[0] = (__bf16)v.x; o[1] = (__bf16)v.y;
        o[2] = (__bf16)v.z; o[3] = (__bf16)v.w;
        *(bf16x4*)(out + i) = o;
    }
}

// ---------------------------------------------------------------------------
// Weight prep: fp32 [R,C] -> bf16 transposed [C,R]. 32x32 LDS tiles.
// ---------------------------------------------------------------------------
__global__ __launch_bounds__(256) void transpose_f32_bf16(
    const float* __restrict__ in, __bf16* __restrict__ out, int R, int C) {
    __shared__ float tile[32][33];
    const int tx = threadIdx.x & 31;
    const int ty = threadIdx.x >> 5;          // 8 rows per pass
    const int c0 = blockIdx.x * 32;
    const int r0 = blockIdx.y * 32;
    #pragma unroll
    for (int i = 0; i < 32; i += 8)
        tile[ty + i][tx] = in[(size_t)(r0 + ty + i) * C + c0 + tx];
    __syncthreads();
    #pragma unroll
    for (int i = 0; i < 32; i += 8)
        out[(size_t)(c0 + ty + i) * R + r0 + tx] = (__bf16)tile[tx][ty + i];
}

// ---------------------------------------------------------------------------
// bf16 GEMM: C[M,N] = A[M,K] * Bt[N,K]^T + bias, optional ReLU.
// Block 128x128, BK=32, 8 waves each owning 32x64 (2x4 WMMA tiles, 8 WMMAs
// per K-step). Both tiles stage to LDS as [row][k] with +8 bf16 padding, so
// A- and B-fragments are each two contiguous 16B LDS reads matching the
// 16-bit 16x32 WMMA lane layout (m = lane&15, half = lane>>4).
// Async path: GLOBAL_LOAD_ASYNC_TO_LDS_B128, depth-2 double buffer,
// s_wait_asynccnt pipelining. Fallback: sync b128 loads + prefetch.
// MODE 0: row-major [M,N] (F32OUT picks fp32/bf16)
// MODE 1: [b][h][s][d] bf16 (Q/K head layout), S=2048, H=16, dk=64
// MODE 2: [b][h][d][s] bf16 (V transposed for flash PV)
// ---------------------------------------------------------------------------
template<int MODE, bool F32OUT, bool RELU>
__global__ __launch_bounds__(256) void gemm_bf16_wmma(
    const __bf16* __restrict__ A, const __bf16* __restrict__ Bt,
    const float* __restrict__ bias, void* __restrict__ Out,
    int M, int N, int K)
{
    constexpr int BK  = 32;
    constexpr int LDP = 40;                    // padded stride (bf16 elems)

    const int tid  = threadIdx.x;
    const int lane = tid & 31;
    const int wave = tid >> 5;
    const int wm   = wave >> 1;                // 0..3 : 32-row sub-tile
    const int wn   = wave & 1;                 // 0..1 : 64-col sub-tile
    const int m16  = lane & 15;
    const int half = lane >> 4;
    const int rowBase = blockIdx.x * 128;
    const int colBase = blockIdx.y * 128;

    v8f acc[2][4] = {};

#if USE_ASYNC_LDS
    __shared__ __bf16 sA[2][128 * LDP];
    __shared__ __bf16 sB[2][128 * LDP];

    auto issue = [&](int buf, int k0) {
        #pragma unroll
        for (int i = 0; i < 2; ++i) {
            int s  = tid + i * 256;
            int r  = s >> 2;
            int ks = (s & 3) * 8;
            __builtin_amdgcn_global_load_async_to_lds_b128(
                as_global_v4i(A + (size_t)(rowBase + r) * K + k0 + ks),
                as_local_v4i(&sA[buf][r * LDP + ks]), 0, 0);
            __builtin_amdgcn_global_load_async_to_lds_b128(
                as_global_v4i(Bt + (size_t)(colBase + r) * K + k0 + ks),
                as_local_v4i(&sB[buf][r * LDP + ks]), 0, 0);
        }
    };

    issue(0, 0);
    if (BK < K) issue(1, BK);
    int pb = 0;
    for (int k0 = 0; k0 < K; k0 += BK) {
        if (k0 + BK < K) __builtin_amdgcn_s_wait_asynccnt(4);  // oldest group done
        else             __builtin_amdgcn_s_wait_asynccnt(0);
        __syncthreads();

        Frag a[2], b[4];
        #pragma unroll
        for (int mi = 0; mi < 2; ++mi) {
            int r = wm * 32 + mi * 16 + m16;
            a[mi].h[0] = *(const bf16x8*)&sA[pb][r * LDP + half * 8];
            a[mi].h[1] = *(const bf16x8*)&sA[pb][r * LDP + 16 + half * 8];
        }
        #pragma unroll
        for (int ni = 0; ni < 4; ++ni) {
            int c = wn * 64 + ni * 16 + m16;
            b[ni].h[0] = *(const bf16x8*)&sB[pb][c * LDP + half * 8];
            b[ni].h[1] = *(const bf16x8*)&sB[pb][c * LDP + 16 + half * 8];
        }
        #pragma unroll
        for (int mi = 0; mi < 2; ++mi)
            #pragma unroll
            for (int ni = 0; ni < 4; ++ni)
                acc[mi][ni] = wmma_bf16(a[mi].v, b[ni].v, acc[mi][ni]);

        __syncthreads();
        if (k0 + 2 * BK < K) issue(pb, k0 + 2 * BK);
        pb ^= 1;
    }
#else
    __shared__ __bf16 sA[128 * LDP];
    __shared__ __bf16 sB[128 * LDP];

    for (int k0 = 0; k0 < K; k0 += BK) {
        #pragma unroll
        for (int i = 0; i < 2; ++i) {
            int s  = tid + i * 256;
            int r  = s >> 2;
            int ks = (s & 3) * 8;
            *(bf16x8*)(&sA[r * LDP + ks]) =
                *(const bf16x8*)(A + (size_t)(rowBase + r) * K + k0 + ks);
            *(bf16x8*)(&sB[r * LDP + ks]) =
                *(const bf16x8*)(Bt + (size_t)(colBase + r) * K + k0 + ks);
        }
        if (k0 + BK < K) {                     // gfx1250 global_prefetch_b8
            __builtin_prefetch(A  + (size_t)(rowBase + (tid >> 2)) * K + k0 + BK, 0, 1);
            __builtin_prefetch(Bt + (size_t)(colBase + (tid >> 2)) * K + k0 + BK, 0, 1);
        }
        __syncthreads();

        Frag a[2], b[4];
        #pragma unroll
        for (int mi = 0; mi < 2; ++mi) {
            int r = wm * 32 + mi * 16 + m16;
            a[mi].h[0] = *(const bf16x8*)&sA[r * LDP + half * 8];
            a[mi].h[1] = *(const bf16x8*)&sA[r * LDP + 16 + half * 8];
        }
        #pragma unroll
        for (int ni = 0; ni < 4; ++ni) {
            int c = wn * 64 + ni * 16 + m16;
            b[ni].h[0] = *(const bf16x8*)&sB[c * LDP + half * 8];
            b[ni].h[1] = *(const bf16x8*)&sB[c * LDP + 16 + half * 8];
        }
        #pragma unroll
        for (int mi = 0; mi < 2; ++mi)
            #pragma unroll
            for (int ni = 0; ni < 4; ++ni)
                acc[mi][ni] = wmma_bf16(a[mi].v, b[ni].v, acc[mi][ni]);
        __syncthreads();
    }
#endif

    // ---- epilogue: bias (+ReLU) + layout-specific store ----
    #pragma unroll
    for (int mi = 0; mi < 2; ++mi) {
        #pragma unroll
        for (int ni = 0; ni < 4; ++ni) {
            int col = colBase + wn * 64 + ni * 16 + m16;
            float bv = bias[col];
            #pragma unroll
            for (int v = 0; v < 8; ++v) {
                int row = rowBase + wm * 32 + mi * 16 + half * 8 + v;
                float val = acc[mi][ni][v] + bv;
                if (RELU) val = fmaxf(val, 0.0f);
                if (MODE == 0) {
                    if (F32OUT) ((float*)Out)[(size_t)row * N + col] = val;
                    else        ((__bf16*)Out)[(size_t)row * N + col] = (__bf16)val;
                } else {
                    int bb = row >> 11, ss = row & 2047;   // S = 2048
                    int hh = col >> 6,  dd = col & 63;     // dk = 64
                    size_t idx = (MODE == 1)
                        ? ((((size_t)bb * 16 + hh) * 2048 + ss) * 64 + dd)
                        : ((((size_t)bb * 16 + hh) * 64 + dd) * 2048 + ss);
                    ((__bf16*)Out)[idx] = (__bf16)val;
                }
            }
        }
    }
}

// ---------------------------------------------------------------------------
// Flash attention (dk=64, S=Skv=2048, scale=1/8). One wave per 16-row Q tile.
// Q,K in [B,H,S,64] bf16; V transposed [B,H,64,S] bf16; out [B,S,1024] bf16.
// Scores: 2 WMMAs (k split 32+32); PV: 4 WMMAs per 32-key step.
// Online softmax row stats per-lane for rows half*8+v (C-matrix layout);
// reductions via shfl_xor over the 16-lane half.
// ---------------------------------------------------------------------------
__global__ __launch_bounds__(128) void flash_attn_wmma(
    const __bf16* __restrict__ Q, const __bf16* __restrict__ K,
    const __bf16* __restrict__ Vt, __bf16* __restrict__ Out)
{
    __shared__ __bf16 sP[4][16 * 40];          // per-wave P staging (16x32, pad)
    const int lane = threadIdx.x & 31;
    const int wave = threadIdx.x >> 5;
    const int g  = blockIdx.x * 4 + wave;      // global q-tile id
    const int bh = g >> 7;                     // 128 q-tiles per (b,h)
    const int qt = g & 127;
    const int m16  = lane & 15;
    const int half = lane >> 4;

    const __bf16* q  = Q  + ((size_t)bh * 2048 + qt * 16) * 64;
    const __bf16* kp = K  + (size_t)bh * 2048 * 64;
    const __bf16* vt = Vt + (size_t)bh * 64 * 2048;

    Frag aq[2];
    #pragma unroll
    for (int c = 0; c < 2; ++c) {
        aq[c].h[0] = *(const bf16x8*)(q + m16 * 64 + c * 32 + half * 8);
        aq[c].h[1] = *(const bf16x8*)(q + m16 * 64 + c * 32 + 16 + half * 8);
    }

    float mrow[8], lrow[8];
    v8f o[4] = {};
    #pragma unroll
    for (int v = 0; v < 8; ++v) { mrow[v] = -3.0e38f; lrow[v] = 0.0f; }

    __bf16* lp = sP[wave];

    for (int j = 0; j < 2048; j += 32) {
        v8f st[2];
        #pragma unroll
        for (int t = 0; t < 2; ++t) {
            const __bf16* kr = kp + (size_t)(j + t * 16 + m16) * 64;
            Frag b0, b1;
            b0.h[0] = *(const bf16x8*)(kr + half * 8);
            b0.h[1] = *(const bf16x8*)(kr + 16 + half * 8);
            b1.h[0] = *(const bf16x8*)(kr + 32 + half * 8);
            b1.h[1] = *(const bf16x8*)(kr + 48 + half * 8);
            v8f s = {};
            s = wmma_bf16(aq[0].v, b0.v, s);
            s = wmma_bf16(aq[1].v, b1.v, s);
            st[t] = s;
        }
        float alpha[8];
        #pragma unroll
        for (int v = 0; v < 8; ++v) {
            float a0 = st[0][v] * 0.125f;
            float a1 = st[1][v] * 0.125f;
            float cm = fmaxf(a0, a1);
            #pragma unroll
            for (int off = 1; off < 16; off <<= 1)
                cm = fmaxf(cm, __shfl_xor(cm, off, 32));
            float mnew = fmaxf(mrow[v], cm);
            float p0 = __expf(a0 - mnew);
            float p1 = __expf(a1 - mnew);
            float rs = p0 + p1;
            #pragma unroll
            for (int off = 1; off < 16; off <<= 1)
                rs += __shfl_xor(rs, off, 32);
            alpha[v] = __expf(mrow[v] - mnew);
            lrow[v]  = lrow[v] * alpha[v] + rs;
            mrow[v]  = mnew;
            st[0][v] = p0;
            st[1][v] = p1;
        }
        #pragma unroll
        for (int nt = 0; nt < 4; ++nt)
            #pragma unroll
            for (int v = 0; v < 8; ++v)
                o[nt][v] *= alpha[v];

        // P (C layout) -> LDS -> A-fragment layout
        #pragma unroll
        for (int t = 0; t < 2; ++t)
            #pragma unroll
            for (int v = 0; v < 8; ++v)
                lp[(half * 8 + v) * 40 + t * 16 + m16] = (__bf16)st[t][v];
        Frag ap;
        ap.h[0] = *(const bf16x8*)&lp[m16 * 40 + half * 8];
        ap.h[1] = *(const bf16x8*)&lp[m16 * 40 + 16 + half * 8];

        #pragma unroll
        for (int nt = 0; nt < 4; ++nt) {
            const __bf16* vp = vt + (size_t)(nt * 16 + m16) * 2048 + j;
            Frag bv;
            bv.h[0] = *(const bf16x8*)(vp + half * 8);
            bv.h[1] = *(const bf16x8*)(vp + 16 + half * 8);
            o[nt] = wmma_bf16(ap.v, bv.v, o[nt]);
        }
    }

    const int bidx = bh >> 4;
    const int hh   = bh & 15;
    #pragma unroll
    for (int nt = 0; nt < 4; ++nt) {
        #pragma unroll
        for (int v = 0; v < 8; ++v) {
            int ss = qt * 16 + half * 8 + v;
            float val = o[nt][v] / lrow[v];
            Out[((size_t)bidx * 2048 + ss) * 1024 + hh * 64 + nt * 16 + m16] =
                (__bf16)val;
        }
    }
}

// ---------------------------------------------------------------------------
// y = LayerNorm(a + r) * g + b ; also emits bf16 copy for the next GEMM.
// One 256-thread block per token; values held in regs so in-place is safe.
// ---------------------------------------------------------------------------
__global__ __launch_bounds__(256) void ln_residual(
    const float* __restrict__ a, const float* __restrict__ r,
    const float* __restrict__ g, const float* __restrict__ be,
    float* __restrict__ y, __bf16* __restrict__ yb)
{
    __shared__ float sred[8];
    const size_t base = (size_t)blockIdx.x * 1024;
    const int lane = threadIdx.x & 31;
    const int wave = threadIdx.x >> 5;

    float xv[4];
    float sum = 0.0f;
    #pragma unroll
    for (int i = 0; i < 4; ++i) {
        int c = threadIdx.x + i * 256;
        xv[i] = a[base + c] + r[base + c];
        sum += xv[i];
    }
    #pragma unroll
    for (int off = 16; off; off >>= 1) sum += __shfl_xor(sum, off, 32);
    if (lane == 0) sred[wave] = sum;
    __syncthreads();
    float tot = 0.0f;
    #pragma unroll
    for (int i = 0; i < 8; ++i) tot += sred[i];
    const float mean = tot * (1.0f / 1024.0f);
    __syncthreads();

    float vs = 0.0f;
    #pragma unroll
    for (int i = 0; i < 4; ++i) { float d = xv[i] - mean; vs += d * d; }
    #pragma unroll
    for (int off = 16; off; off >>= 1) vs += __shfl_xor(vs, off, 32);
    if (lane == 0) sred[wave] = vs;
    __syncthreads();
    float vtot = 0.0f;
    #pragma unroll
    for (int i = 0; i < 8; ++i) vtot += sred[i];
    const float rstd = rsqrtf(vtot * (1.0f / 1024.0f) + 1e-5f);

    #pragma unroll
    for (int i = 0; i < 4; ++i) {
        int c = threadIdx.x + i * 256;
        float o = (xv[i] - mean) * rstd * g[c] + be[c];
        y[base + c]  = o;
        yb[base + c] = (__bf16)o;
    }
}

// ---------------------------------------------------------------------------
// Host launch
// ---------------------------------------------------------------------------
extern "C" void kernel_launch(void* const* d_in, const int* in_sizes, int n_in,
                              void* d_out, int out_size, void* d_ws, size_t ws_size,
                              hipStream_t stream) {
    (void)in_sizes; (void)n_in; (void)out_size; (void)ws_size;
    constexpr int B = 4, S = 2048, D = 1024, DFF = 4096;
    constexpr int NT = B * S;                       // 8192 tokens

    const float* src   = (const float*)d_in[0];
    const float* tgt   = (const float*)d_in[1];
    const float* sa_wq = (const float*)d_in[2];  const float* sa_bq = (const float*)d_in[3];
    const float* sa_wk = (const float*)d_in[4];  const float* sa_bk = (const float*)d_in[5];
    const float* sa_wv = (const float*)d_in[6];  const float* sa_bv = (const float*)d_in[7];
    const float* sa_wo = (const float*)d_in[8];  const float* sa_bo = (const float*)d_in[9];
    const float* ca_wq = (const float*)d_in[10]; const float* ca_bq = (const float*)d_in[11];
    const float* ca_wk = (const float*)d_in[12]; const float* ca_bk = (const float*)d_in[13];
    const float* ca_wv = (const float*)d_in[14]; const float* ca_bv = (const float*)d_in[15];
    const float* ca_wo = (const float*)d_in[16]; const float* ca_bo = (const float*)d_in[17];
    const float* ff_w1 = (const float*)d_in[18]; const float* ff_b1 = (const float*)d_in[19];
    const float* ff_w2 = (const float*)d_in[20]; const float* ff_b2 = (const float*)d_in[21];
    const float* ln1_g = (const float*)d_in[22]; const float* ln1_b = (const float*)d_in[23];
    const float* ln2_g = (const float*)d_in[24]; const float* ln2_b = (const float*)d_in[25];
    const float* ln3_g = (const float*)d_in[26]; const float* ln3_b = (const float*)d_in[27];

    // ---- workspace carve-up (256B aligned) ----
    char* ws = (char*)d_ws;
    size_t off = 0;
    auto alloc = [&](size_t bytes) -> char* {
        char* p = ws + off;
        off += (bytes + 255) & ~(size_t)255;
        return p;
    };
    __bf16* xq_b    = (__bf16*)alloc((size_t)NT * D * 2);   // current query stream (bf16)
    __bf16* xsrc_b  = (__bf16*)alloc((size_t)NT * D * 2);   // encoder memory (bf16)
    __bf16* qh      = (__bf16*)alloc((size_t)NT * D * 2);   // [B,H,S,64]
    __bf16* kh      = (__bf16*)alloc((size_t)NT * D * 2);   // [B,H,S,64]
    __bf16* vtb     = (__bf16*)alloc((size_t)NT * D * 2);   // [B,H,64,S]
    __bf16* attn_b  = (__bf16*)alloc((size_t)NT * D * 2);   // attn out bf16 [NT,D]
    float*  proj_f  = (float*) alloc((size_t)NT * D * 4);   // GEMM fp32 out
    float*  x_f     = (float*) alloc((size_t)NT * D * 4);   // fp32 residual stream
    __bf16* hid_b   = (__bf16*)alloc((size_t)NT * DFF * 2); // FF hidden bf16
    __bf16* w_saq   = (__bf16*)alloc((size_t)D * D * 2);    // all weights: [N,K] bf16
    __bf16* w_sak   = (__bf16*)alloc((size_t)D * D * 2);
    __bf16* w_sav   = (__bf16*)alloc((size_t)D * D * 2);
    __bf16* w_sao   = (__bf16*)alloc((size_t)D * D * 2);
    __bf16* w_caq   = (__bf16*)alloc((size_t)D * D * 2);
    __bf16* w_cak   = (__bf16*)alloc((size_t)D * D * 2);
    __bf16* w_cav   = (__bf16*)alloc((size_t)D * D * 2);
    __bf16* w_cao   = (__bf16*)alloc((size_t)D * D * 2);
    __bf16* w1b     = (__bf16*)alloc((size_t)D * DFF * 2);
    __bf16* w2b     = (__bf16*)alloc((size_t)DFF * D * 2);

    auto cvt = [&](const float* s, __bf16* d, int n) {
        cvt_f32_bf16<<<n / 1024, 256, 0, stream>>>(s, d, n);
    };
    auto wtr = [&](const float* s, __bf16* d, int R, int C) {   // [R,C] -> [C,R] bf16
        transpose_f32_bf16<<<dim3(C / 32, R / 32), 256, 0, stream>>>(s, d, R, C);
    };
    // ---- precision conversion / weight transposition ----
    cvt(tgt, xq_b, NT * D);
    cvt(src, xsrc_b, NT * D);
    wtr(sa_wq, w_saq, D, D); wtr(sa_wk, w_sak, D, D);
    wtr(sa_wv, w_sav, D, D); wtr(sa_wo, w_sao, D, D);
    wtr(ca_wq, w_caq, D, D); wtr(ca_wk, w_cak, D, D);
    wtr(ca_wv, w_cav, D, D); wtr(ca_wo, w_cao, D, D);
    wtr(ff_w1, w1b, D, DFF); wtr(ff_w2, w2b, DFF, D);

    const dim3 blk(256);
    const dim3 gD(NT / 128, D / 128);       // 64 x 8
    const dim3 gFF(NT / 128, DFF / 128);    // 64 x 32
    const int flashBlocks = B * 16 * (S / 16) / 4;  // 2048

    // ---- self-attention ----
    gemm_bf16_wmma<1, false, false><<<gD, blk, 0, stream>>>(xq_b, w_saq, sa_bq, qh,  NT, D, D);
    gemm_bf16_wmma<1, false, false><<<gD, blk, 0, stream>>>(xq_b, w_sak, sa_bk, kh,  NT, D, D);
    gemm_bf16_wmma<2, false, false><<<gD, blk, 0, stream>>>(xq_b, w_sav, sa_bv, vtb, NT, D, D);
    flash_attn_wmma<<<flashBlocks, 128, 0, stream>>>(qh, kh, vtb, attn_b);
    gemm_bf16_wmma<0, true,  false><<<gD, blk, 0, stream>>>(attn_b, w_sao, sa_bo, proj_f, NT, D, D);
    ln_residual<<<NT, 256, 0, stream>>>(tgt, proj_f, ln1_g, ln1_b, x_f, xq_b);

    // ---- cross-attention (Q from decoder stream, K/V from src) ----
    gemm_bf16_wmma<1, false, false><<<gD, blk, 0, stream>>>(xq_b,   w_caq, ca_bq, qh,  NT, D, D);
    gemm_bf16_wmma<1, false, false><<<gD, blk, 0, stream>>>(xsrc_b, w_cak, ca_bk, kh,  NT, D, D);
    gemm_bf16_wmma<2, false, false><<<gD, blk, 0, stream>>>(xsrc_b, w_cav, ca_bv, vtb, NT, D, D);
    flash_attn_wmma<<<flashBlocks, 128, 0, stream>>>(qh, kh, vtb, attn_b);
    gemm_bf16_wmma<0, true,  false><<<gD, blk, 0, stream>>>(attn_b, w_cao, ca_bo, proj_f, NT, D, D);
    ln_residual<<<NT, 256, 0, stream>>>(x_f, proj_f, ln2_g, ln2_b, x_f, xq_b);

    // ---- feed-forward ----
    gemm_bf16_wmma<0, false, true ><<<gFF, blk, 0, stream>>>(xq_b,  w1b, ff_b1, hid_b,  NT, DFF, D);
    gemm_bf16_wmma<0, true,  false><<<gD,  blk, 0, stream>>>(hid_b, w2b, ff_b2, proj_f, NT, D, DFF);
    ln_residual<<<NT, 256, 0, stream>>>(x_f, proj_f, ln3_g, ln3_b, (float*)d_out, xq_b);
}
